// PreciseADR_36352603193983
// MI455X (gfx1250) — matrix operand order repaired and checked by reading, not compiled, not run on hardware
//
#include <hip/hip_runtime.h>
#include <hip/hip_bf16.h>
#include <math.h>
#include <stddef.h>

typedef __attribute__((ext_vector_type(16))) _Float16 v16h;
typedef __attribute__((ext_vector_type(8)))  _Float16 v8h;
typedef __attribute__((ext_vector_type(4)))  _Float16 v4h;
typedef __attribute__((ext_vector_type(8)))  float    v8f;

#define HID 256
#define HEADS 8
#define DH 32

// ---------------------------------------------------------------------------
// Generic GEMM: C[M,N] = act(A[M,K] @ W[K,N] + b[N]), fp32 in/out,
// f16 WMMA with fp32 accumulate.
// Block tile 64(M) x 128(N), K step 32; 8 waves, each wave owns a 2x2 grid of
// 16x16 WMMA tiles -> 4 v_wmma per K step with fragment reuse.
// LDS layouts match the CDNA5 wave32 fragment maps so lanes load contiguous
// 16-byte runs (CDNA5 ISA 7.12.2).
// ---------------------------------------------------------------------------
#define GBM 64
#define GBN 128
#define GBK 32
#define LDP 40   // LDS row stride in halves (80B = 5x16B, keeps 16B alignment)

// A-column permutation: lane lo needs K {0..7,16..23}, lane hi {8..15,24..31}.
// Swapping bits 3<->4 of the K index makes each lane's 16 halves contiguous.
__device__ __forceinline__ int permA(int c) {
  return (c & ~24) | ((c & 8) << 1) | ((c & 16) >> 1);
}

__global__ __launch_bounds__(256)
void gemm_wmma_kernel(const float* __restrict__ A, const float* __restrict__ W,
                      const float* __restrict__ bias, float* __restrict__ C,
                      int M, int N, int K, int act)
{
  __shared__ __align__(16) _Float16 sAp[GBM][LDP];   // [m][permuted k]
  __shared__ __align__(16) _Float16 sBt[GBN][LDP];   // [n][k] (transposed)

  const int m0 = blockIdx.y * GBM;
  const int n0 = blockIdx.x * GBN;
  const int t  = threadIdx.x;
  const int w  = t >> 5;
  const int l  = t & 31;
  const int wm = w >> 2;   // 0..1
  const int wn = w & 3;    // 0..3
  const int ln = l & 15;
  const int lh = l >> 4;

  const bool interiorMN = (m0 + GBM <= M) && (n0 + GBN <= N);
  const bool vec4 = ((K & 3) == 0) && ((N & 3) == 0);

  v8f acc[2][2];
  #pragma unroll
  for (int mi = 0; mi < 2; ++mi)
    #pragma unroll
    for (int ni = 0; ni < 2; ++ni)
      acc[mi][ni] = (v8f){0.f,0.f,0.f,0.f,0.f,0.f,0.f,0.f};

  for (int k0 = 0; k0 < K; k0 += GBK) {
    const bool fast = interiorMN && vec4 && (k0 + GBK <= K);  // block-uniform

    // ---- stage A tile (GBM x GBK), 8 floats/thread ----
    {
      const int ar = t >> 2;          // 0..63
      const int ac = (t & 3) * 8;     // 0,8,16,24
      if (fast) {
        const float* ap = A + (size_t)(m0 + ar) * K + k0 + ac;
        float4 u0 = *(const float4*)(ap);
        float4 u1 = *(const float4*)(ap + 4);
        v8h hv;
        hv[0]=(_Float16)u0.x; hv[1]=(_Float16)u0.y; hv[2]=(_Float16)u0.z; hv[3]=(_Float16)u0.w;
        hv[4]=(_Float16)u1.x; hv[5]=(_Float16)u1.y; hv[6]=(_Float16)u1.z; hv[7]=(_Float16)u1.w;
        *(v8h*)&sAp[ar][permA(ac)] = hv;   // group-contiguous under permA
      } else {
        // branch-free: clamp address, load unconditionally, select 0
        v8h hv;
        #pragma unroll
        for (int j = 0; j < 8; ++j) {
          int gm = m0 + ar, gk = k0 + ac + j;
          bool ok = (gm < M) && (gk < K);
          int gmc = gm < M ? gm : (M - 1);
          int gkc = gk < K ? gk : (K - 1);
          float v = A[(size_t)gmc * K + gkc];
          hv[j] = (_Float16)(ok ? v : 0.f);
        }
        *(v8h*)&sAp[ar][permA(ac)] = hv;
      }
    }
    // ---- stage B tile (GBK x GBN) transposed, 16 floats/thread ----
    {
      const int cc = t & 127;         // n within tile
      const int rh = t >> 7;          // 0..1 (k half)
      v8h h0, h1;
      #pragma unroll
      for (int j = 0; j < 16; ++j) {
        int gk = k0 + rh * 16 + j, gn = n0 + cc;
        bool ok = (gk < K) && (gn < N);
        int gkc = gk < K ? gk : (K - 1);
        int gnc = gn < N ? gn : (N - 1);
        float v = W[(size_t)gkc * N + gnc];    // coalesced across lanes
        _Float16 hv = (_Float16)(ok ? v : 0.f);
        if (j < 8) h0[j] = hv; else h1[j - 8] = hv;
      }
      *(v8h*)&sBt[cc][rh * 16]     = h0;
      *(v8h*)&sBt[cc][rh * 16 + 8] = h1;
    }
    __syncthreads();

    // ---- fragments: contiguous 16-byte runs per lane ----
    v16h af[2], bf[2];
    #pragma unroll
    for (int mi = 0; mi < 2; ++mi) {
      int row = wm * 32 + mi * 16 + ln;
      v8h a0 = *(const v8h*)&sAp[row][lh * 16];
      v8h a1 = *(const v8h*)&sAp[row][lh * 16 + 8];
      af[mi] = __builtin_shufflevector(a0, a1, 0,1,2,3,4,5,6,7,8,9,10,11,12,13,14,15);
    }
    #pragma unroll
    for (int ni = 0; ni < 2; ++ni) {
      int col = wn * 32 + ni * 16 + ln;
      v8h b0 = *(const v8h*)&sBt[col][lh * 8];
      v8h b1 = *(const v8h*)&sBt[col][lh * 8 + 16];
      bf[ni] = __builtin_shufflevector(b0, b1, 0,1,2,3,4,5,6,7,8,9,10,11,12,13,14,15);
    }
    #pragma unroll
    for (int mi = 0; mi < 2; ++mi)
      #pragma unroll
      for (int ni = 0; ni < 2; ++ni)
        acc[mi][ni] = __builtin_amdgcn_wmma_f32_16x16x32_f16(
            false, af[mi], false, bf[ni], (short)0, acc[mi][ni], false, false);
    __syncthreads();
  }

  // ---- epilogue: lane N = l&15, VGPR r -> M = (l>>4)*8 + r ----
  #pragma unroll
  for (int ni = 0; ni < 2; ++ni) {
    int col = n0 + wn * 32 + ni * 16 + ln;
    float bv = (col < N) ? bias[col] : 0.f;
    #pragma unroll
    for (int mi = 0; mi < 2; ++mi) {
      #pragma unroll
      for (int r = 0; r < 8; ++r) {
        int m = m0 + wm * 32 + mi * 16 + lh * 8 + r;
        if (m < M && col < N) {
          float v = acc[mi][ni][r] + bv;
          if (act == 1)      v = fmaxf(v, 0.f);
          else if (act == 2) v = 0.5f * v * (1.f + erff(v * 0.70710678118654752f));
          C[(size_t)m * N + col] = v;
        }
      }
    }
  }
}

// ---------------------------------------------------------------------------
// Per-head relation projection: Y[n, h*32+e] = sum_d X[n, h*32+d] * R[h,d,e]
// R[head] staged to LDS transposed once per block; X loaded via float4.
// One wave -> one 16x16 tile, single K=32 WMMA.
// ---------------------------------------------------------------------------
__global__ __launch_bounds__(256)
void headproj_kernel(const float* __restrict__ X, const float* __restrict__ R,
                     float* __restrict__ Y, int M)
{
  __shared__ __align__(16) _Float16 sRt[DH][LDP];   // [e][d] transposed
  const int t = threadIdx.x;
  const int ntile = blockIdx.x;        // 0..15
  const int head  = ntile >> 1;

  { // stage R[head] (32x32) transposed: 4 values/thread
    const int cc = t & 31;             // e
    const int rh = t >> 5;             // 0..7 -> d rows rh*4..rh*4+3
    const float* rp = R + (size_t)head * (DH * DH);
    v4h hv;
    #pragma unroll
    for (int j = 0; j < 4; ++j)
      hv[j] = (_Float16)rp[(rh * 4 + j) * DH + cc];
    *(v4h*)&sRt[cc][rh * 4] = hv;
  }
  __syncthreads();

  const int w = t >> 5;
  const int l = t & 31;
  const int ln = l & 15;
  const int lh = l >> 4;
  const int mtile = blockIdx.y * 8 + w;
  const int m0 = mtile * 16;
  if (m0 < M) {                         // wave-uniform, no barrier inside
    int row = m0 + ln;
    bool ok = row < M;
    int rowc = ok ? row : (M - 1);
    const float* xp = X + (size_t)rowc * HID + head * DH;
    int kb = lh * 8;
    float4 u0 = *(const float4*)(xp + kb);
    float4 u1 = *(const float4*)(xp + kb + 4);
    float4 u2 = *(const float4*)(xp + 16 + kb);
    float4 u3 = *(const float4*)(xp + 20 + kb);
    if (!ok) { u0 = u1 = u2 = u3 = make_float4(0.f, 0.f, 0.f, 0.f); }
    v16h af;
    af[0]=(_Float16)u0.x;  af[1]=(_Float16)u0.y;  af[2]=(_Float16)u0.z;  af[3]=(_Float16)u0.w;
    af[4]=(_Float16)u1.x;  af[5]=(_Float16)u1.y;  af[6]=(_Float16)u1.z;  af[7]=(_Float16)u1.w;
    af[8]=(_Float16)u2.x;  af[9]=(_Float16)u2.y;  af[10]=(_Float16)u2.z; af[11]=(_Float16)u2.w;
    af[12]=(_Float16)u3.x; af[13]=(_Float16)u3.y; af[14]=(_Float16)u3.z; af[15]=(_Float16)u3.w;

    int col16 = (ntile & 1) * 16 + ln;
    v8h b0 = *(const v8h*)&sRt[col16][lh * 8];
    v8h b1 = *(const v8h*)&sRt[col16][lh * 8 + 16];
    v16h bf = __builtin_shufflevector(b0, b1, 0,1,2,3,4,5,6,7,8,9,10,11,12,13,14,15);

    v8f acc = {0.f,0.f,0.f,0.f,0.f,0.f,0.f,0.f};
    acc = __builtin_amdgcn_wmma_f32_16x16x32_f16(false, af, false, bf,
                                                 (short)0, acc, false, false);
    int col = ntile * 16 + ln;
    #pragma unroll
    for (int r = 0; r < 8; ++r) {
      int m = m0 + lh * 8 + r;
      if (m < M) Y[(size_t)m * HID + col] = acc[r];
    }
  }
}

// ---------------------------------------------------------------------------
// Edge / scatter-softmax kernels
// ---------------------------------------------------------------------------
__device__ inline unsigned f2ord(float f) {
  unsigned u = __float_as_uint(f);
  return (u & 0x80000000u) ? ~u : (u | 0x80000000u);
}
__device__ inline float ord2f(unsigned u) {
  return __uint_as_float((u & 0x80000000u) ? (u & 0x7fffffffu) : ~u);
}

__global__ void edge_alpha_kernel(const int* __restrict__ ei, int E,
                                  const float* __restrict__ krel,
                                  const float* __restrict__ q,
                                  const float* __restrict__ prel,
                                  float* __restrict__ alpha,
                                  unsigned* __restrict__ amax)
{
  int t = blockIdx.x * blockDim.x + threadIdx.x;
  if (t >= E * HEADS) return;
  int e = t >> 3, h = t & 7;
  int si = ei[e], di = ei[E + e];
  const float* kp = krel + (size_t)si * HID + h * DH;
  const float* qp = q    + (size_t)di * HID + h * DH;
  float s = 0.f;
  #pragma unroll
  for (int d = 0; d < DH; ++d) s += kp[d] * qp[d];
  s *= prel[h] * 0.17677669529663687f;   // 1/sqrt(32)
  alpha[t] = s;
  atomicMax(&amax[(size_t)di * HEADS + h], f2ord(s));
}

__global__ void amax_fin_kernel(unsigned* __restrict__ amax, int n)
{
  int t = blockIdx.x * blockDim.x + threadIdx.x;
  if (t >= n) return;
  float m = ord2f(amax[t]);
  if (!isfinite(m)) m = 0.f;
  ((float*)amax)[t] = m;
}

__global__ void edge_expsum_kernel(const int* __restrict__ ei, int E,
                                   float* __restrict__ alpha,
                                   const float* __restrict__ amaxf,
                                   float* __restrict__ asum)
{
  int t = blockIdx.x * blockDim.x + threadIdx.x;
  if (t >= E * HEADS) return;
  int e = t >> 3, h = t & 7;
  int di = ei[E + e];
  float ae = expf(alpha[t] - amaxf[(size_t)di * HEADS + h]);
  alpha[t] = ae;
  atomicAdd(&asum[(size_t)di * HEADS + h], ae);
}

__global__ void edge_agg_kernel(const int* __restrict__ ei, int E,
                                const float* __restrict__ ae,
                                const float* __restrict__ asum,
                                const float* __restrict__ vrel,
                                float* __restrict__ agg)
{
  int t = blockIdx.x * blockDim.x + threadIdx.x;
  if (t >= E * HEADS) return;
  int e = t >> 3, h = t & 7;
  int si = ei[e], di = ei[E + e];
  float wgt = ae[t] / (asum[(size_t)di * HEADS + h] + 1e-16f);
  const float* vp = vrel + (size_t)si * HID + h * DH;
  float* ap = agg + (size_t)di * HID + h * DH;
  #pragma unroll
  for (int d = 0; d < DH; ++d) atomicAdd(&ap[d], wgt * vp[d]);
}

// ---------------------------------------------------------------------------
// Elementwise helpers
// ---------------------------------------------------------------------------
__global__ void fillf_kernel(float* p, size_t n, float v)
{ size_t t = (size_t)blockIdx.x * blockDim.x + threadIdx.x; if (t < n) p[t] = v; }

__global__ void fillu_kernel(unsigned* p, size_t n, unsigned v)
{ size_t t = (size_t)blockIdx.x * blockDim.x + threadIdx.x; if (t < n) p[t] = v; }

__global__ void gelu_kernel(const float* __restrict__ x, float* __restrict__ y, size_t n)
{
  size_t t = (size_t)blockIdx.x * blockDim.x + threadIdx.x;
  if (t >= n) return;
  float v = x[t];
  y[t] = 0.5f * v * (1.f + erff(v * 0.70710678118654752f));
}

__global__ void skipmix_relu_kernel(float* __restrict__ h, const float* __restrict__ o,
                                    const float* __restrict__ skip, size_t n)
{
  size_t t = (size_t)blockIdx.x * blockDim.x + threadIdx.x;
  if (t >= n) return;
  float s = 1.f / (1.f + expf(-skip[0]));
  h[t] = fmaxf(s * o[t] + (1.f - s) * h[t], 0.f);
}

// ---------------------------------------------------------------------------
// Host orchestration
// ---------------------------------------------------------------------------
extern "C" void kernel_launch(void* const* d_in, const int* in_sizes, int n_in,
                              void* d_out, int out_size, void* d_ws, size_t ws_size,
                              hipStream_t stream)
{
  (void)in_sizes; (void)n_in; (void)out_size; (void)ws_size;
  auto F = [&](int i) { return (const float*)d_in[i]; };
  auto I = [&](int i) { return (const int*)d_in[i]; };

  // JAX pytree order: node types sorted [adr, disease, drug, patient]
  const int NN[4]  = {1000, 1200, 1500, 20000};
  const int DIN[4] = {800, 900, 1200, 7};
  const int RSRC[6] = {0, 1, 2, 3, 3, 3};
  const int RDST[6] = {3, 3, 3, 0, 1, 2};
  const int RE[6]   = {40000, 60000, 100000, 40000, 60000, 100000};
  // Input indices: edges 0..5; aug.b=6 aug.w=7; per layer l: base = 8+54*l with
  //   a_rel[r]=+r, k[nt]{b,w}=+6+2nt/+7+2nt, m_rel[r]=+14+r, o[nt]=+20+2nt/+21+2nt,
  //   p_rel[r]=+28+r, q[nt]=+34+2nt/+35+2nt, skip[nt]=+42+nt, v[nt]=+46+2nt/+47+2nt;
  // pred.b=170 pred.w=171; proj[nt]{b,w}=172+2nt/173+2nt; x[nt]=180+nt.

  float* ws = (float*)d_ws;
  size_t off = 0;
  size_t off_h[4], off_k[4], off_q[4], off_v[4], off_agg[4], off_amax[4], off_asum[4], off_alpha[6];
  for (int nt = 0; nt < 4; ++nt) { off_h[nt]   = off; off += (size_t)NN[nt] * HID; }
  for (int nt = 0; nt < 4; ++nt) { off_k[nt]   = off; off += (size_t)NN[nt] * HID; }
  for (int nt = 0; nt < 4; ++nt) { off_q[nt]   = off; off += (size_t)NN[nt] * HID; }
  for (int nt = 0; nt < 4; ++nt) { off_v[nt]   = off; off += (size_t)NN[nt] * HID; }
  for (int nt = 0; nt < 4; ++nt) { off_agg[nt] = off; off += (size_t)NN[nt] * HID; }
  size_t off_krel = off; off += (size_t)20000 * HID;   // reused as gelu tmp
  size_t off_vrel = off; off += (size_t)20000 * HID;   // reused as O tmp
  for (int nt = 0; nt < 4; ++nt) { off_amax[nt] = off; off += (size_t)NN[nt] * HEADS; }
  for (int nt = 0; nt < 4; ++nt) { off_asum[nt] = off; off += (size_t)NN[nt] * HEADS; }
  for (int r = 0; r < 6; ++r)    { off_alpha[r] = off; off += (size_t)RE[r] * HEADS; }

  auto gemm = [&](const float* A, const float* W, const float* b, float* C,
                  int M, int N, int K, int act) {
    dim3 g((N + GBN - 1) / GBN, (M + GBM - 1) / GBM);
    gemm_wmma_kernel<<<g, 256, 0, stream>>>(A, W, b, C, M, N, K, act);
  };
  auto ew = [&](size_t n) { return dim3((unsigned)((n + 255) / 256)); };

  // 1) Node feature projection + ReLU
  for (int nt = 0; nt < 4; ++nt)
    gemm(F(180 + nt), F(173 + 2 * nt), F(172 + 2 * nt), ws + off_h[nt],
         NN[nt], HID, DIN[nt], 1);

  // 2) HGT layers
  for (int l = 0; l < 3; ++l) {
    int LB = 8 + 54 * l;
    for (int nt = 0; nt < 4; ++nt) {
      gemm(ws + off_h[nt], F(LB + 7 + 2 * nt),  F(LB + 6 + 2 * nt),  ws + off_k[nt], NN[nt], HID, HID, 0);
      gemm(ws + off_h[nt], F(LB + 35 + 2 * nt), F(LB + 34 + 2 * nt), ws + off_q[nt], NN[nt], HID, HID, 0);
      gemm(ws + off_h[nt], F(LB + 47 + 2 * nt), F(LB + 46 + 2 * nt), ws + off_v[nt], NN[nt], HID, HID, 0);
      size_t nh = (size_t)NN[nt] * HEADS;
      fillu_kernel<<<ew(nh), 256, 0, stream>>>((unsigned*)(ws + off_amax[nt]), nh, 0x007FFFFFu); // ord(-inf)
      fillf_kernel<<<ew(nh), 256, 0, stream>>>(ws + off_asum[nt], nh, 0.f);
      size_t na = (size_t)NN[nt] * HID;
      fillf_kernel<<<ew(na), 256, 0, stream>>>(ws + off_agg[nt], na, 0.f);
    }
    for (int r = 0; r < 6; ++r) {
      int s = RSRC[r], d = RDST[r];
      headproj_kernel<<<dim3(16, (NN[s] + 127) / 128), 256, 0, stream>>>(
          ws + off_k[s], F(LB + r), ws + off_krel, NN[s]);
      edge_alpha_kernel<<<ew((size_t)RE[r] * HEADS), 256, 0, stream>>>(
          I(r), RE[r], ws + off_krel, ws + off_q[d], F(LB + 28 + r),
          ws + off_alpha[r], (unsigned*)(ws + off_amax[d]));
    }
    for (int nt = 0; nt < 4; ++nt)
      amax_fin_kernel<<<ew((size_t)NN[nt] * HEADS), 256, 0, stream>>>(
          (unsigned*)(ws + off_amax[nt]), NN[nt] * HEADS);
    for (int r = 0; r < 6; ++r)
      edge_expsum_kernel<<<ew((size_t)RE[r] * HEADS), 256, 0, stream>>>(
          I(r), RE[r], ws + off_alpha[r], ws + off_amax[RDST[r]], ws + off_asum[RDST[r]]);
    for (int r = 0; r < 6; ++r) {
      int s = RSRC[r], d = RDST[r];
      headproj_kernel<<<dim3(16, (NN[s] + 127) / 128), 256, 0, stream>>>(
          ws + off_v[s], F(LB + 14 + r), ws + off_vrel, NN[s]);
      edge_agg_kernel<<<ew((size_t)RE[r] * HEADS), 256, 0, stream>>>(
          I(r), RE[r], ws + off_alpha[r], ws + off_asum[d], ws + off_vrel, ws + off_agg[d]);
    }
    for (int nt = 0; nt < 4; ++nt) {
      size_t na = (size_t)NN[nt] * HID;
      gelu_kernel<<<ew(na), 256, 0, stream>>>(ws + off_agg[nt], ws + off_krel, na);
      gemm(ws + off_krel, F(LB + 21 + 2 * nt), F(LB + 20 + 2 * nt), ws + off_vrel,
           NN[nt], HID, HID, 0);
      skipmix_relu_kernel<<<ew(na), 256, 0, stream>>>(ws + off_h[nt], ws + off_vrel,
                                                      F(LB + 42 + nt), na);
    }
  }

  // 3) Outputs: [logits (20000x1000), h_patient (20000x256), h_aug (20000x256)]
  float* out        = (float*)d_out;
  float* out_logits = out;
  float* out_hp     = out + (size_t)20000 * 1000;
  float* out_aug    = out_hp + (size_t)20000 * HID;
  (void)hipMemcpyAsync(out_hp, ws + off_h[3], (size_t)20000 * HID * sizeof(float),
                       hipMemcpyDeviceToDevice, stream);
  gemm(ws + off_h[3], F(7),   F(6),   out_aug,    20000, HID,  HID, 0);  // aug linear
  gemm(out_aug,       F(171), F(170), out_logits, 20000, 1000, HID, 0);  // predictor
}
